// EMAAdaptiveDecayModel_21784074125427
// MI455X (gfx1250) — compile-verified, matrix operand order checked by reading, not run on hardware
//
#include <hip/hip_runtime.h>
#include <hip/hip_bf16.h>

// ---------------------------------------------------------------------------
// CDNA5 (gfx1250, wave32) implementation.
// All large GEMMs use v_wmma_f32_16x16x32_bf16 (bf16 in / f32 accumulate).
// Weight tiles stream to LDS via global_load_async_to_lds_b128 when available.
// f32->bf16 staging uses packed v_cvt_pk_bf16_f32 when available.
// ---------------------------------------------------------------------------

typedef __attribute__((ext_vector_type(16))) __bf16 v16bf;
typedef __attribute__((ext_vector_type(2)))  __bf16 v2bf;
typedef __attribute__((ext_vector_type(8)))  float  v8f;
typedef int vsi4 __attribute__((vector_size(16)));   // async builtin's b128 unit

#define B_  32
#define T_  512
#define HID_ 256
#define HEADS_ 8
#define DK_ 32

#if defined(__gfx1250__) && __has_builtin(__builtin_amdgcn_global_load_async_to_lds_b128)
#define USE_ASYNC_LDS 1
#else
#define USE_ASYNC_LDS 0
#endif

__device__ __forceinline__ unsigned short f2bf(float f) {
    unsigned int u = __float_as_uint(f);
    u += 0x7FFFu + ((u >> 16) & 1u);        // round-to-nearest-even truncate
    return (unsigned short)(u >> 16);
}
__device__ __forceinline__ unsigned pack2(float a, float b) {
#if defined(__gfx1250__) && __has_builtin(__builtin_amdgcn_cvt_pk_bf16_f32)
    union { v2bf v; unsigned u; } c;
    c.v = __builtin_amdgcn_cvt_pk_bf16_f32(a, b);   // v_cvt_pk_bf16_f32 (RNE)
    return c.u;
#else
    return (unsigned)f2bf(a) | ((unsigned)f2bf(b) << 16);
#endif
}

// convert 16 consecutive f32 -> 16 bf16 (32B) into LDS
__device__ __forceinline__ void stageA16(const float* __restrict__ src,
                                         unsigned short* __restrict__ dst) {
    const float4* s = (const float4*)src;
    float4 f0 = s[0], f1 = s[1], f2 = s[2], f3 = s[3];
    uint4 lo = make_uint4(pack2(f0.x, f0.y), pack2(f0.z, f0.w),
                          pack2(f1.x, f1.y), pack2(f1.z, f1.w));
    uint4 hi = make_uint4(pack2(f2.x, f2.y), pack2(f2.z, f2.w),
                          pack2(f3.x, f3.y), pack2(f3.z, f3.w));
    ((uint4*)dst)[0] = lo;
    ((uint4*)dst)[1] = hi;
}
__device__ __forceinline__ void stageZero16(unsigned short* __restrict__ dst) {
    uint4 z = make_uint4(0u, 0u, 0u, 0u);
    ((uint4*)dst)[0] = z;
    ((uint4*)dst)[1] = z;
}
// copy 16 bf16 (32B) global -> LDS; async-to-LDS on gfx1250 toolchains that have it
__device__ __forceinline__ void stageCopy16(const unsigned short* __restrict__ g,
                                            unsigned short* __restrict__ l) {
#if USE_ASYNC_LDS
    __builtin_amdgcn_global_load_async_to_lds_b128(
        (__attribute__((address_space(1))) vsi4*)(uintptr_t)g,
        (__attribute__((address_space(3))) vsi4*)(unsigned)(uintptr_t)l, 0, 0);
    __builtin_amdgcn_global_load_async_to_lds_b128(
        (__attribute__((address_space(1))) vsi4*)(uintptr_t)(g + 8),
        (__attribute__((address_space(3))) vsi4*)(unsigned)(uintptr_t)(l + 8), 0, 0);
#else
    ((uint4*)l)[0] = ((const uint4*)g)[0];
    ((uint4*)l)[1] = ((const uint4*)g)[1];
#endif
}
__device__ __forceinline__ void wait_async() {
#if USE_ASYNC_LDS
#if __has_builtin(__builtin_amdgcn_s_wait_asynccnt)
    __builtin_amdgcn_s_wait_asynccnt(0);
#else
    asm volatile("s_wait_asynccnt 0x0" ::: "memory");
#endif
#endif
}

union FragBF { v16bf v; unsigned int u[8]; };

// Load one 16x32 bf16 operand fragment from an LDS tile (rows of 32 halfs).
// A-matrix layout (ISA 7.12.2): lane L -> row M=L%16; K base = (L<16?0:8);
// v0..3 hold K=base..base+7, v4..7 hold K=base+16..base+23 (two 16B chunks).
// B fragments use identical math on W^T tiles (lane -> column N).
__device__ __forceinline__ v16bf ldsFrag(const unsigned short* base) {
    int lane = threadIdx.x & 31;
    int r    = lane & 15;
    int kb   = (lane >> 4) << 3;            // 0 or 8
    const unsigned int* p = (const unsigned int*)base;
    int d0 = r * 16 + (kb >> 1);            // 16 dwords per 32-half row
    FragBF f;
#pragma unroll
    for (int i = 0; i < 4; ++i) f.u[i]     = p[d0 + i];
#pragma unroll
    for (int i = 0; i < 4; ++i) f.u[4 + i] = p[d0 + 8 + i];
    return f.v;
}

// ---------------------------------------------------------------------------
// Weight packing (runs every call; cheap, deterministic)
// ---------------------------------------------------------------------------
__global__ void pack_wT(const float* __restrict__ W, unsigned short* __restrict__ Wt,
                        int K, int N, int KP) {
    int idx = blockIdx.x * 256 + threadIdx.x;
    if (idx >= N * KP) return;
    int n = idx / KP, k = idx % KP;
    Wt[idx] = (k < K) ? f2bf(W[(size_t)k * N + n]) : (unsigned short)0;
}

// conv weight (O,I,KS=3) -> bf16 (tap,O,I)
__global__ void pack_conv(const float* __restrict__ W, unsigned short* __restrict__ Wp) {
    int idx = blockIdx.x * 256 + threadIdx.x;
    if (idx >= 3 * 256 * 256) return;
    int tap = idx / 65536, rem = idx % 65536;
    int o = rem / 256, i = rem % 256;
    Wp[idx] = f2bf(W[((size_t)o * 256 + i) * 3 + tap]);
}

// ---------------------------------------------------------------------------
// Generic WMMA GEMM: C(MxN) = act(A(f32 dense MxKP) @ W^T_bf16(NxKP) + bias)[+res]
// block = 128 threads (4 waves); block tile 64x64; wave tile 16x64.
// ---------------------------------------------------------------------------
__global__ __launch_bounds__(128)
void gemm_wmma(const float* __restrict__ A, int lda,
               const unsigned short* __restrict__ Bt, int KP,
               const float* __restrict__ bias,
               const float* __restrict__ residual,
               float* __restrict__ C, int ldc, int relu)
{
    __shared__ __align__(16) unsigned short As[64 * 32];
    __shared__ __align__(16) unsigned short Bs[64 * 32];
    const int m0 = blockIdx.y * 64, n0 = blockIdx.x * 64;
    const int tid = threadIdx.x, wave = tid >> 5;
    const int sr = tid >> 1, sc = (tid & 1) * 16;   // staging: row, 16-col chunk

    v8f z = {0.f, 0.f, 0.f, 0.f, 0.f, 0.f, 0.f, 0.f};
    v8f acc[4] = {z, z, z, z};

    for (int k0 = 0; k0 < KP; k0 += 32) {
        const float* aSrc = A + (size_t)(m0 + sr) * lda + k0 + sc;
        stageA16(aSrc, &As[sr * 32 + sc]);
        stageCopy16(Bt + (size_t)(n0 + sr) * KP + k0 + sc, &Bs[sr * 32 + sc]);
        if (k0 + 32 < KP) __builtin_prefetch(aSrc + 32);
        wait_async();
        __syncthreads();
        v16bf a = ldsFrag(As + wave * 16 * 32);
#pragma unroll
        for (int nt = 0; nt < 4; ++nt) {
            v16bf b = ldsFrag(Bs + nt * 16 * 32);
            acc[nt] = __builtin_amdgcn_wmma_f32_16x16x32_bf16(
                false, a, false, b, (short)0, acc[nt], false, false);
        }
        __syncthreads();
    }
    int lane = tid & 31, nlo = lane & 15, mAdd = (lane >> 4) * 8;
#pragma unroll
    for (int nt = 0; nt < 4; ++nt) {
        int n = n0 + nt * 16 + nlo;
        float bv = bias ? bias[n] : 0.f;
#pragma unroll
        for (int vi = 0; vi < 8; ++vi) {
            int m = m0 + wave * 16 + mAdd + vi;
            float val = acc[nt][vi] + bv;
            if (relu) val = fmaxf(val, 0.f);
            if (residual) val += residual[(size_t)m * ldc + n];
            C[(size_t)m * ldc + n] = val;
        }
    }
}

// ---------------------------------------------------------------------------
// Causal dilated conv (KS=3) as 3 shifted-A WMMA GEMMs accumulated together.
// y[b,t,o] = relu(bias[o] + sum_tap sum_i w[o,i,tap]*x[b, t-(2-tap)*dil, i])[+res]
// X layout (B*T, 256). Tile rows never cross batch boundary (512 % 64 == 0).
// ---------------------------------------------------------------------------
__global__ __launch_bounds__(128)
void conv_wmma(const float* __restrict__ X,
               const unsigned short* __restrict__ Wp,  // (3,256,256) bf16 W^T-per-tap
               const float* __restrict__ bias,
               const float* __restrict__ residual,
               float* __restrict__ Y, int dil)
{
    __shared__ __align__(16) unsigned short As[64 * 32];
    __shared__ __align__(16) unsigned short Bs[64 * 32];
    const int m0 = blockIdx.y * 64, n0 = blockIdx.x * 64;
    const int b = m0 >> 9, tbase = m0 & 511;
    const int tid = threadIdx.x, wave = tid >> 5;
    const int sr = tid >> 1, sc = (tid & 1) * 16;

    v8f z = {0.f, 0.f, 0.f, 0.f, 0.f, 0.f, 0.f, 0.f};
    v8f acc[4] = {z, z, z, z};

    for (int tap = 0; tap < 3; ++tap) {
        int t = tbase + sr - (2 - tap) * dil;   // staging row's source time
        for (int k0 = 0; k0 < 256; k0 += 32) {
            if (t >= 0) {
                const float* aSrc = X + ((size_t)b * T_ + t) * 256 + k0 + sc;
                stageA16(aSrc, &As[sr * 32 + sc]);
                if (k0 + 32 < 256) __builtin_prefetch(aSrc + 32);
            } else {
                stageZero16(&As[sr * 32 + sc]);
            }
            stageCopy16(Wp + ((size_t)tap * 256 + n0 + sr) * 256 + k0 + sc,
                        &Bs[sr * 32 + sc]);
            wait_async();
            __syncthreads();
            v16bf a = ldsFrag(As + wave * 16 * 32);
#pragma unroll
            for (int nt = 0; nt < 4; ++nt) {
                v16bf bb = ldsFrag(Bs + nt * 16 * 32);
                acc[nt] = __builtin_amdgcn_wmma_f32_16x16x32_bf16(
                    false, a, false, bb, (short)0, acc[nt], false, false);
            }
            __syncthreads();
        }
    }
    int lane = tid & 31, nlo = lane & 15, mAdd = (lane >> 4) * 8;
#pragma unroll
    for (int nt = 0; nt < 4; ++nt) {
        int n = n0 + nt * 16 + nlo;
        float bv = bias[n];
#pragma unroll
        for (int vi = 0; vi < 8; ++vi) {
            int m = m0 + wave * 16 + mAdd + vi;
            float val = fmaxf(acc[nt][vi] + bv, 0.f);
            if (residual) val += residual[(size_t)m * 256 + n];
            Y[(size_t)m * 256 + n] = val;
        }
    }
}

// ---------------------------------------------------------------------------
// Attention scores: scores[b,h,t,s] = (q.k)/sqrt(32) - lam[b,h,s]*dt[b,t,s],
// masked to -inf for s >= len[b].  One K=32 WMMA step per tile.
// ---------------------------------------------------------------------------
__global__ __launch_bounds__(128)
void scores_wmma(const float* __restrict__ q, const float* __restrict__ k,
                 const float* __restrict__ lam, const float* __restrict__ dt,
                 const int* __restrict__ lengths, float* __restrict__ attn)
{
    __shared__ __align__(16) unsigned short As[64 * 32];
    __shared__ __align__(16) unsigned short Bs[64 * 32];
    const int bh = blockIdx.z, b = bh >> 3, hd = bh & 7;
    const int t0 = blockIdx.y * 64, s0 = blockIdx.x * 64;
    const int tid = threadIdx.x, wave = tid >> 5;
    const int sr = tid >> 1, sc = (tid & 1) * 16;

    stageA16(q + ((size_t)b * T_ + t0 + sr) * HID_ + hd * DK_ + sc, &As[sr * 32 + sc]);
    stageA16(k + ((size_t)b * T_ + s0 + sr) * HID_ + hd * DK_ + sc, &Bs[sr * 32 + sc]);
    __syncthreads();

    v8f z = {0.f, 0.f, 0.f, 0.f, 0.f, 0.f, 0.f, 0.f};
    v8f acc[4] = {z, z, z, z};
    v16bf a = ldsFrag(As + wave * 16 * 32);
#pragma unroll
    for (int nt = 0; nt < 4; ++nt) {
        v16bf bb = ldsFrag(Bs + nt * 16 * 32);
        acc[nt] = __builtin_amdgcn_wmma_f32_16x16x32_bf16(
            false, a, false, bb, (short)0, acc[nt], false, false);
    }

    const float scale = 0.17677669529663687f;  // 1/sqrt(32)
    int lane = tid & 31, nlo = lane & 15, mAdd = (lane >> 4) * 8;
    int len = lengths[b];
#pragma unroll
    for (int nt = 0; nt < 4; ++nt) {
        int s = s0 + nt * 16 + nlo;
        float lm = lam[((size_t)b * T_ + s) * HEADS_ + hd];
#pragma unroll
        for (int vi = 0; vi < 8; ++vi) {
            int t = t0 + wave * 16 + mAdd + vi;
            float sc2;
            if (s < len)
                sc2 = acc[nt][vi] * scale - lm * dt[((size_t)b * T_ + t) * T_ + s];
            else
                sc2 = -__builtin_inff();
            attn[((size_t)bh * T_ + t) * T_ + s] = sc2;
        }
    }
}

// In-place row softmax over 512 keys (block = 256 threads, 2 elems/thread)
__global__ __launch_bounds__(256)
void softmax_rows(float* __restrict__ attn) {
    __shared__ float red[256];
    float* p = attn + (size_t)blockIdx.x * T_;
    int tid = threadIdx.x;
    float a = p[tid], bv = p[tid + 256];
    red[tid] = fmaxf(a, bv);
    __syncthreads();
    for (int s = 128; s > 0; s >>= 1) {
        if (tid < s) red[tid] = fmaxf(red[tid], red[tid + s]);
        __syncthreads();
    }
    float mx = red[0];
    __syncthreads();
    float e0 = expf(a - mx), e1 = expf(bv - mx);
    red[tid] = e0 + e1;
    __syncthreads();
    for (int s = 128; s > 0; s >>= 1) {
        if (tid < s) red[tid] += red[tid + s];
        __syncthreads();
    }
    float inv = 1.f / red[0];
    p[tid] = e0 * inv;
    p[tid + 256] = e1 * inv;
}

// ctx[b,t,h*32+n] = sum_s attn[b,h,t,s] * v[b,s,h*32+n]; K=512 (16 WMMA steps)
__global__ __launch_bounds__(128)
void ctx_wmma(const float* __restrict__ attn, const float* __restrict__ v,
              float* __restrict__ ctx)
{
    __shared__ __align__(16) unsigned short As[64 * 32];
    __shared__ __align__(16) unsigned short Bs[32 * 32];
    const int bh = blockIdx.y, b = bh >> 3, hd = bh & 7;
    const int t0 = blockIdx.x * 64;
    const int tid = threadIdx.x, wave = tid >> 5;
    const int sr = tid >> 1, sc = (tid & 1) * 16;

    v8f z = {0.f, 0.f, 0.f, 0.f, 0.f, 0.f, 0.f, 0.f};
    v8f acc[2] = {z, z};

    for (int k0 = 0; k0 < T_; k0 += 32) {
        const float* aSrc = attn + ((size_t)bh * T_ + t0 + sr) * T_ + k0 + sc;
        stageA16(aSrc, &As[sr * 32 + sc]);
        if (k0 + 32 < T_) __builtin_prefetch(aSrc + 32);
        // transpose-stage V^T: Bs[n*32+c] = v[(b*T+k0+c)*256 + hd*32 + n]
#pragma unroll
        for (int i = 0; i < 8; ++i) {
            int idx = i * 128 + tid;
            int c = idx >> 5, n = idx & 31;     // consecutive tid -> consecutive n
            Bs[n * 32 + c] = f2bf(v[((size_t)b * T_ + k0 + c) * HID_ + hd * DK_ + n]);
        }
        __syncthreads();
        v16bf a = ldsFrag(As + wave * 16 * 32);
#pragma unroll
        for (int nt = 0; nt < 2; ++nt) {
            v16bf bb = ldsFrag(Bs + nt * 16 * 32);
            acc[nt] = __builtin_amdgcn_wmma_f32_16x16x32_bf16(
                false, a, false, bb, (short)0, acc[nt], false, false);
        }
        __syncthreads();
    }
    int lane = tid & 31, nlo = lane & 15, mAdd = (lane >> 4) * 8;
#pragma unroll
    for (int nt = 0; nt < 2; ++nt) {
        int n = nt * 16 + nlo;
#pragma unroll
        for (int vi = 0; vi < 8; ++vi) {
            int t = t0 + wave * 16 + mAdd + vi;
            ctx[((size_t)b * T_ + t) * HID_ + hd * DK_ + n] = acc[nt][vi];
        }
    }
}

// ---------------------------------------------------------------------------
// Pointwise / scan / head kernels
// ---------------------------------------------------------------------------
__device__ __forceinline__ float gelu_f(float x) {
    return 0.5f * x * (1.f + erff(x * 0.70710678118654752f));
}

// X is (B*T, 128): 120 real features + 8 zero-pad (keeps GEMM A dense)
__global__ __launch_bounds__(128)
void build_X(const int* __restrict__ sid, const int* __restrict__ stid,
             const int* __restrict__ vtid, const float* __restrict__ num,
             const float* __restrict__ nmask, const float* __restrict__ tf,
             const float* __restrict__ semb, const float* __restrict__ stemb,
             const float* __restrict__ vtemb,
             const float* __restrict__ num_w, const float* __restrict__ num_b,
             const float* __restrict__ time_w, const float* __restrict__ time_b,
             float* __restrict__ X)
{
    int row = blockIdx.x, j = threadIdx.x;
    float val = 0.f;
    if (j < 64)       val = semb[(size_t)sid[row] * 64 + j];
    else if (j < 80)  val = stemb[(size_t)stid[row] * 16 + (j - 64)];
    else if (j < 88)  val = vtemb[(size_t)vtid[row] * 8 + (j - 80)];
    else if (j < 104) {
        int c = j - 88;
        val = gelu_f(num[row] * num_w[c] + nmask[row] * num_w[16 + c] + num_b[c]);
    } else if (j < 120) {
        int c = j - 104;
        float x = time_b[c];
#pragma unroll
        for (int i = 0; i < 4; ++i) x += tf[(size_t)row * 4 + i] * time_w[i * 16 + c];
        val = gelu_f(x);
    }
    X[(size_t)row * 128 + j] = val;
}

__global__ __launch_bounds__(256)
void ema_kernel(const float* __restrict__ h, const int* __restrict__ lengths,
                float* __restrict__ out) {
    int b = blockIdx.x, d = threadIdx.x;
    int len = lengths[b];
    const float* hp = h + (size_t)b * T_ * HID_ + d;
    float* op = out + (size_t)b * T_ * HID_ + d;
    float prev = hp[0];
    op[0] = prev;
    for (int t = 1; t < T_; ++t) {
        float x = hp[(size_t)t * HID_];
        float nv = 0.8f * prev + 0.2f * x;
        prev = (t < len) ? nv : prev;
        op[(size_t)t * HID_] = prev;
    }
}

__global__ __launch_bounds__(256)
void lam_kernel(const float* __restrict__ cond, const float* __restrict__ l1w,
                const float* __restrict__ l1b, const float* __restrict__ l2w,
                const float* __restrict__ l2b, float* __restrict__ lam) {
    __shared__ float hid[256];
    int row = blockIdx.x, j = threadIdx.x;
    const float* c = cond + (size_t)row * 8;
    float x = l1b[j];
#pragma unroll
    for (int i = 0; i < 8; ++i) x += c[i] * l1w[i * 256 + j];
    hid[j] = fmaxf(x, 0.f);
    __syncthreads();
    if (j < 8) {
        float s = l2b[j];
        for (int i = 0; i < 256; ++i) s += hid[i] * l2w[i * 8 + j];
        lam[(size_t)row * 8 + j] = (s > 20.f) ? s : log1pf(expf(s));
    }
}

__global__ __launch_bounds__(256)
void pool_kernel(const float* __restrict__ seq, const int* __restrict__ lengths,
                 float* __restrict__ pooled) {
    int b = blockIdx.x, d = threadIdx.x;
    int len = lengths[b];
    const float* p = seq + (size_t)b * T_ * HID_ + d;
    float s = 0.f;
    for (int t = 0; t < len; ++t) s += p[(size_t)t * HID_];
    int dn = len < 1 ? 1 : len;
    pooled[b * HID_ + d] = s / (float)dn;
}

__global__ __launch_bounds__(256)
void head_kernel(const float* __restrict__ pooled,
                 const float* __restrict__ c1w, const float* __restrict__ c1b,
                 const float* __restrict__ c2w, const float* __restrict__ c2b,
                 const float* __restrict__ c3w, const float* __restrict__ c3b,
                 float* __restrict__ logits) {
    __shared__ float pz[256], z1[256], z2[128];
    int b = blockIdx.x, j = threadIdx.x;
    pz[j] = pooled[b * 256 + j];
    __syncthreads();
    float s = c1b[j];
    for (int i = 0; i < 256; ++i) s += pz[i] * c1w[i * 256 + j];
    z1[j] = fmaxf(s, 0.f);
    __syncthreads();
    if (j < 128) {
        float s2 = c2b[j];
        for (int i = 0; i < 256; ++i) s2 += z1[i] * c2w[i * 128 + j];
        z2[j] = fmaxf(s2, 0.f);
    }
    __syncthreads();
    if (j < 5) {
        float s3 = c3b[j];
        for (int i = 0; i < 128; ++i) s3 += z2[i] * c3w[i * 5 + j];
        logits[b * 5 + j] = s3;
    }
}

// ---------------------------------------------------------------------------
// Host launch
// ---------------------------------------------------------------------------
static inline char* ws_align(char*& cur, size_t bytes) {
    uintptr_t p = (uintptr_t)cur;
    p = (p + 255) & ~(uintptr_t)255;
    char* r = (char*)p;
    cur = r + bytes;
    return r;
}

extern "C" void kernel_launch(void* const* d_in, const int* in_sizes, int n_in,
                              void* d_out, int out_size, void* d_ws, size_t ws_size,
                              hipStream_t stream) {
    const int*   sensor_ids = (const int*)d_in[0];
    const int*   state_ids  = (const int*)d_in[1];
    const int*   vt_ids     = (const int*)d_in[2];
    const float* numeric    = (const float*)d_in[3];
    const float* nmask      = (const float*)d_in[4];
    const float* timef      = (const float*)d_in[5];
    const float* cond       = (const float*)d_in[6];
    const float* delta_t    = (const float*)d_in[7];
    const int*   lengths    = (const int*)d_in[8];
    const float* sensor_emb = (const float*)d_in[9];
    const float* state_emb  = (const float*)d_in[10];
    const float* vt_emb     = (const float*)d_in[11];
    const float* num_w = (const float*)d_in[12];
    const float* num_b = (const float*)d_in[13];
    const float* time_w = (const float*)d_in[14];
    const float* time_b = (const float*)d_in[15];
    const float* feat_w = (const float*)d_in[16];
    const float* feat_b = (const float*)d_in[17];
    const float* tcn_w[6] = {(const float*)d_in[18], (const float*)d_in[20],
                             (const float*)d_in[22], (const float*)d_in[24],
                             (const float*)d_in[26], (const float*)d_in[28]};
    const float* tcn_b[6] = {(const float*)d_in[19], (const float*)d_in[21],
                             (const float*)d_in[23], (const float*)d_in[25],
                             (const float*)d_in[27], (const float*)d_in[29]};
    const float* q_w = (const float*)d_in[30]; const float* q_b = (const float*)d_in[31];
    const float* k_w = (const float*)d_in[32]; const float* k_b = (const float*)d_in[33];
    const float* v_w = (const float*)d_in[34]; const float* v_b = (const float*)d_in[35];
    const float* o_w = (const float*)d_in[36]; const float* o_b = (const float*)d_in[37];
    const float* l1_w = (const float*)d_in[38]; const float* l1_b = (const float*)d_in[39];
    const float* l2_w = (const float*)d_in[40]; const float* l2_b = (const float*)d_in[41];
    const float* c1_w = (const float*)d_in[42]; const float* c1_b = (const float*)d_in[43];
    const float* c2_w = (const float*)d_in[44]; const float* c2_b = (const float*)d_in[45];
    const float* c3_w = (const float*)d_in[46]; const float* c3_b = (const float*)d_in[47];

    const int M = B_ * T_;                      // 16384
    // output tuple offsets (floats): logits, seq_out, pooled, attn, h_ema
    float* out = (float*)d_out;
    float* out_logits = out;
    float* out_seq    = out + 160;
    float* out_pool   = out_seq + (size_t)M * HID_;
    float* out_attn   = out_pool + (size_t)B_ * HID_;
    float* out_ema    = out_attn + (size_t)B_ * HEADS_ * T_ * T_;

    // workspace carve-out
    char* cur = (char*)d_ws;
    float* X   = (float*)ws_align(cur, (size_t)M * 128 * 4);   // padded K=128
    float* H   = (float*)ws_align(cur, (size_t)M * HID_ * 4);
    float* S1  = (float*)ws_align(cur, (size_t)M * HID_ * 4);
    float* S2  = (float*)ws_align(cur, (size_t)M * HID_ * 4);
    float* Q   = (float*)ws_align(cur, (size_t)M * HID_ * 4);
    float* Kb  = (float*)ws_align(cur, (size_t)M * HID_ * 4);
    float* Vb  = (float*)ws_align(cur, (size_t)M * HID_ * 4);
    float* CTX = (float*)ws_align(cur, (size_t)M * HID_ * 4);
    float* LAM = (float*)ws_align(cur, (size_t)M * HEADS_ * 4);
    unsigned short* featT = (unsigned short*)ws_align(cur, 256 * 128 * 2);
    unsigned short* qT = (unsigned short*)ws_align(cur, 256 * 256 * 2);
    unsigned short* kT = (unsigned short*)ws_align(cur, 256 * 256 * 2);
    unsigned short* vT = (unsigned short*)ws_align(cur, 256 * 256 * 2);
    unsigned short* oT = (unsigned short*)ws_align(cur, 256 * 256 * 2);
    unsigned short* convP[6];
    for (int i = 0; i < 6; ++i)
        convP[i] = (unsigned short*)ws_align(cur, 3 * 256 * 256 * 2);

    // 1) pack weights to bf16 (transposed / per-tap)
    pack_wT<<<(256 * 128 + 255) / 256, 256, 0, stream>>>(feat_w, featT, 120, 256, 128);
    pack_wT<<<256, 256, 0, stream>>>(q_w, qT, 256, 256, 256);
    pack_wT<<<256, 256, 0, stream>>>(k_w, kT, 256, 256, 256);
    pack_wT<<<256, 256, 0, stream>>>(v_w, vT, 256, 256, 256);
    pack_wT<<<256, 256, 0, stream>>>(o_w, oT, 256, 256, 256);
    for (int i = 0; i < 6; ++i)
        pack_conv<<<(3 * 65536 + 255) / 256, 256, 0, stream>>>(tcn_w[i], convP[i]);

    // 2) features -> X (B*T,128 padded)
    build_X<<<M, 128, 0, stream>>>(sensor_ids, state_ids, vt_ids, numeric, nmask,
                                   timef, sensor_emb, state_emb, vt_emb,
                                   num_w, num_b, time_w, time_b, X);

    // 3) h = X @ feat_w + feat_b  (WMMA)
    gemm_wmma<<<dim3(4, M / 64), 128, 0, stream>>>(X, 128, featT, 128,
                                                   feat_b, nullptr, H, 256, 0);

    // 4) h_ema (sequential scan) -> output
    ema_kernel<<<B_, 256, 0, stream>>>(H, lengths, out_ema);

    // 5) TCN: 3 residual layers of 2 dilated causal convs (WMMA)
    conv_wmma<<<dim3(4, M / 64), 128, 0, stream>>>(H,  convP[0], tcn_b[0], nullptr, S1, 1);
    conv_wmma<<<dim3(4, M / 64), 128, 0, stream>>>(S1, convP[1], tcn_b[1], H,       S2, 1);
    conv_wmma<<<dim3(4, M / 64), 128, 0, stream>>>(S2, convP[2], tcn_b[2], nullptr, H,  2);
    conv_wmma<<<dim3(4, M / 64), 128, 0, stream>>>(H,  convP[3], tcn_b[3], S2,      S1, 2);
    conv_wmma<<<dim3(4, M / 64), 128, 0, stream>>>(S1, convP[4], tcn_b[4], nullptr, S2, 4);
    conv_wmma<<<dim3(4, M / 64), 128, 0, stream>>>(S2, convP[5], tcn_b[5], S1,      H,  4);
    // final sequence features now in H

    // 6) Q,K,V projections (WMMA)
    gemm_wmma<<<dim3(4, M / 64), 128, 0, stream>>>(H, 256, qT, 256, q_b, nullptr, Q, 256, 0);
    gemm_wmma<<<dim3(4, M / 64), 128, 0, stream>>>(H, 256, kT, 256, k_b, nullptr, Kb, 256, 0);
    gemm_wmma<<<dim3(4, M / 64), 128, 0, stream>>>(H, 256, vT, 256, v_b, nullptr, Vb, 256, 0);

    // 7) adaptive decay rates lam[b,s,h]
    lam_kernel<<<M, 256, 0, stream>>>(cond, l1_w, l1_b, l2_w, l2_b, LAM);

    // 8) scores with decay + mask (WMMA) -> attn region, then softmax in place
    scores_wmma<<<dim3(8, 8, B_ * HEADS_), 128, 0, stream>>>(Q, Kb, LAM, delta_t,
                                                             lengths, out_attn);
    softmax_rows<<<B_ * HEADS_ * T_, 256, 0, stream>>>(out_attn);

    // 9) context = attn @ V (WMMA, attn f32 -> bf16 on the fly)
    ctx_wmma<<<dim3(8, B_ * HEADS_), 128, 0, stream>>>(out_attn, Vb, CTX);

    // 10) seq_out = ctx @ o_w + o_b (WMMA) -> output region
    gemm_wmma<<<dim3(4, M / 64), 128, 0, stream>>>(CTX, 256, oT, 256, o_b,
                                                   nullptr, out_seq, 256, 0);

    // 11) masked mean pooling + classifier head
    pool_kernel<<<B_, 256, 0, stream>>>(out_seq, lengths, out_pool);
    head_kernel<<<B_, 256, 0, stream>>>(out_pool, c1_w, c1_b, c2_w, c2_b,
                                        c3_w, c3_b, out_logits);
}
